// Net_19542101197583
// MI455X (gfx1250) — matrix operand order repaired and checked by reading, not compiled, not run on hardware
//
#include <hip/hip_runtime.h>
#include <hip/hip_bf16.h>

#define NN    8192      // nodes
#define NE    32768     // edges
#define FDIM  16
#define DIM   64
#define BTY   4
#define RNK   512
#define KH    33        // 32 hidden channels + 1 bias slot
#define PCOLS (KH*DIM)  // 2112
#define KPAD  32        // padded K for the input-layer GEMM

typedef __attribute__((ext_vector_type(16))) __bf16 bf16x16;
typedef __attribute__((ext_vector_type(8)))  __bf16 bf16x8;
typedef __attribute__((ext_vector_type(2)))  __bf16 bf16x2;
typedef __attribute__((ext_vector_type(8)))  float  f32x8;
typedef __attribute__((ext_vector_type(2)))  float  f32x2;

// ---------------------------------------------------------------------------
// bf16 WMMA GEMM:  C[M,N] = act( A[M,K] @ B[N,K]^T + addend*rowscale + bias )
// Preconditions: M%16==0, K%32==0, A/B bf16 row-major, rows 16B aligned.
// Each wave computes NT adjacent 16x16 tiles (A-fragment reuse).
// Hot loop: 2 b128 loads (A) + 2*NT b128 loads (B) + NT wmma, no branches.
// ---------------------------------------------------------------------------
template<int NT>
__global__ __launch_bounds__(256) void wmma_gemm(
    const __bf16* __restrict__ A, const __bf16* __restrict__ B,
    float* __restrict__ Cf, __bf16* __restrict__ Cb,
    const float* __restrict__ bias, const float* __restrict__ addend,
    const float* __restrict__ rowscale,
    int M, int N, int K, int lda, int ldb, int ldc, int relu)
{
    const int lane = threadIdx.x & 31;
    const int half = lane >> 4;           // wave32 lane groups 0-15 / 16-31
    const int l16  = lane & 15;
    const int wid  = blockIdx.x * (blockDim.x >> 5) + (threadIdx.x >> 5);

    const int tilesM = M >> 4;
    const int grpN   = (N + NT * 16 - 1) / (NT * 16);
    if (wid >= tilesM * grpN) return;     // wave-uniform exit: EXEC all-1 for WMMA
    const int mBase  = (wid / grpN) << 4;
    const int nBase0 = (wid % grpN) * NT * 16;

    f32x8 acc[NT];
#pragma unroll
    for (int t = 0; t < NT; ++t) acc[t] = (f32x8){0,0,0,0,0,0,0,0};

    const __bf16* __restrict__ Arow = A + (size_t)(mBase + l16) * lda;
    const __bf16* __restrict__ Brow[NT];
#pragma unroll
    for (int t = 0; t < NT; ++t) {
        int c = nBase0 + t * 16 + l16;
        if (c >= N) c = N - 1;            // clamp: garbage col never stored
        Brow[t] = B + (size_t)c * ldb;
    }

    for (int k0 = 0; k0 < K; k0 += 32) {
        // A 16x32 frag: lanes<16: K k0+[0..7],k0+[16..23]; lanes>=16: +8
        bf16x8 a0 = *(const bf16x8*)(Arow + k0 + half * 8);
        bf16x8 a1 = *(const bf16x8*)(Arow + k0 + 16 + half * 8);
        bf16x16 af = __builtin_shufflevector(a0, a1,
            0,1,2,3,4,5,6,7,8,9,10,11,12,13,14,15);
#pragma unroll
        for (int t = 0; t < NT; ++t) {
            // B 32x16 frag: lane col = l16, K rows k0+half*8+[0..7], k0+16+half*8+[0..7]
            bf16x8 b0 = *(const bf16x8*)(Brow[t] + k0 + half * 8);
            bf16x8 b1 = *(const bf16x8*)(Brow[t] + k0 + 16 + half * 8);
            bf16x16 bfv = __builtin_shufflevector(b0, b1,
                0,1,2,3,4,5,6,7,8,9,10,11,12,13,14,15);
            acc[t] = __builtin_amdgcn_wmma_f32_16x16x32_bf16(
                false, af, false, bfv, (short)0, acc[t], false, false);
        }
    }

#pragma unroll
    for (int t = 0; t < NT; ++t) {
        const int col = nBase0 + t * 16 + l16;
        if (col >= N) continue;
#pragma unroll
        for (int r = 0; r < 8; ++r) {
            const int row = mBase + r + half * 8;   // C layout: VGPR r -> M=r(+8)
            float v = acc[t][r];
            if (addend) v += addend[(size_t)row * N + col] * (rowscale ? rowscale[row] : 1.0f);
            if (bias)   v += bias[col];
            if (relu)   v = fmaxf(v, 0.0f);
            if (Cf) Cf[(size_t)row * ldc + col] = v;
            if (Cb) Cb[(size_t)row * ldc + col] = (__bf16)v;
        }
    }
}

// ---------------------------------------------------------------------------
__global__ void zero_f32(float* p, int n) {
    int i = blockIdx.x * blockDim.x + threadIdx.x;
    if (i < n) p[i] = 0.0f;
}

__global__ void count_deg(const int* __restrict__ dst, float* deg, int e) {
    int i = blockIdx.x * blockDim.x + threadIdx.x;
    if (i < e) atomicAdd(&deg[dst[i]], 1.0f);
}

__global__ void finalize_invdeg(float* invdeg, const float* __restrict__ deg, int n) {
    int i = blockIdx.x * blockDim.x + threadIdx.x;
    if (i < n) invdeg[i] = 1.0f / fmaxf(deg[i], 1.0f);
}

// generic f32 -> bf16 convert ([N,K] row-major weights used directly as B)
__global__ void cvt_bf16(const float* __restrict__ in, __bf16* __restrict__ out, int n) {
    int i = blockIdx.x * blockDim.x + threadIdx.x;
    if (i < n) out[i] = (__bf16)in[i];
}

// pad K 16->32 with zeros, f32 -> bf16 (for x and W0)
__global__ void pad32_bf16(const float* __restrict__ in, __bf16* __restrict__ out, int rows) {
    int i = blockIdx.x * blockDim.x + threadIdx.x;
    if (i >= rows * KPAD) return;
    int r = i / KPAD, k = i % KPAD;
    out[i] = (__bf16)((k < FDIM) ? in[r * FDIM + k] : 0.0f);
}

// rootT[n,k] = root[k,n]  (so root GEMM can use the B^T path)
__global__ void rootT_bf16(const float* __restrict__ root, __bf16* __restrict__ out) {
    int i = blockIdx.x * blockDim.x + threadIdx.x;
    if (i >= DIM * DIM) return;
    int n = i / DIM, k = i % DIM;
    out[i] = (__bf16)root[k * DIM + n];
}

// TmatT[(k*64+o), i] = nnW2[i*64+o, k] (k<32); k==32 row carries nnb2  -> [PCOLS,64]
__global__ void tmatT_bf16(const float* __restrict__ w2, const float* __restrict__ b2,
                           __bf16* __restrict__ out) {
    int idx = blockIdx.x * blockDim.x + threadIdx.x;
    if (idx >= PCOLS * DIM) return;
    int c = idx / DIM, i = idx % DIM;
    int k = c / DIM, o = c % DIM;
    float v = (k < 32) ? w2[(size_t)(i * DIM + o) * 32 + k] : b2[i * DIM + o];
    out[idx] = (__bf16)v;
}

// Ut[t][n, r] = U[t, r, n]   -> B^T form [64,512] per bond type
__global__ void uT_bf16(const float* __restrict__ U, __bf16* __restrict__ out) {
    int idx = blockIdx.x * blockDim.x + threadIdx.x;
    if (idx >= BTY * DIM * RNK) return;
    int t = idx / (DIM * RNK), rem = idx % (DIM * RNK);
    int n = rem / RNK, r = rem % RNK;
    out[idx] = (__bf16)U[(size_t)t * RNK * DIM + (size_t)r * DIM + n];
}

// hidden[e,j] = relu(edge_attr[e,:4] @ nnW1[j,:4] + nnb1[j])
__global__ void hidden_kernel(const float* __restrict__ ea, const float* __restrict__ w1,
                              const float* __restrict__ b1, float* __restrict__ hid, int e) {
    int i = blockIdx.x * blockDim.x + threadIdx.x;
    if (i >= e * 32) return;
    int eid = i >> 5, j = i & 31;
    float v = b1[j];
    for (int t = 0; t < BTY; ++t) v += ea[eid * BTY + t] * w1[j * BTY + t];
    hid[i] = fmaxf(v, 0.0f);
}

// one wave per edge; lane owns cols 2*lane, 2*lane+1 (paired b32 bf16 loads)
__global__ __launch_bounds__(256) void edge_msg(
    const int* __restrict__ src, const int* __restrict__ dst,
    const float* __restrict__ hid, const __bf16* __restrict__ P,
    float* __restrict__ agg, int e)
{
    int w = (blockIdx.x * blockDim.x + threadIdx.x) >> 5;
    int lane = threadIdx.x & 31;
    if (w >= e) return;
    int s = src[w], d = dst[w];
    const __bf16* __restrict__ Pr = P + (size_t)s * PCOLS + 2 * lane;
    const float*  __restrict__ hr = hid + (size_t)w * 32;
    bf16x2 pb = *(const bf16x2*)(Pr + 32 * DIM);      // bias slot (hidden==1)
    float a0 = (float)pb[0], a1 = (float)pb[1];
#pragma unroll 8
    for (int k = 0; k < 32; ++k) {
        float hk = hr[k];
        bf16x2 pv = *(const bf16x2*)(Pr + k * DIM);
        a0 += hk * (float)pv[0];
        a1 += hk * (float)pv[1];
    }
    atomicAdd(&agg[(size_t)d * DIM + 2 * lane],     a0);
    atomicAdd(&agg[(size_t)d * DIM + 2 * lane + 1], a1);
}

// GRU cell (torch gate order r,z,n); writes f32 state + bf16 shadow
__global__ void gru_kernel(const float* __restrict__ gi, const float* __restrict__ gh,
                           float* __restrict__ state, __bf16* __restrict__ state_bf, int n) {
    int i = blockIdx.x * blockDim.x + threadIdx.x;
    if (i >= n * DIM) return;
    int nid = i / DIM, j = i % DIM;
    const float* a = gi + (size_t)nid * 3 * DIM;
    const float* b = gh + (size_t)nid * 3 * DIM;
    float r  = 1.0f / (1.0f + expf(-(a[j] + b[j])));
    float z  = 1.0f / (1.0f + expf(-(a[DIM + j] + b[DIM + j])));
    float nn = tanhf(a[2 * DIM + j] + r * b[2 * DIM + j]);
    float v  = (1.0f - z) * nn + z * state[i];
    state[i] = v;
    state_bf[i] = (__bf16)v;
}

// out_combine: masked upscale of out_edges (x[:,0]==2) else node state
__global__ void combine_kernel(const float* __restrict__ x, const float* __restrict__ oe,
                               const float* __restrict__ wup, const float* __restrict__ bup,
                               const float* __restrict__ state, float* __restrict__ hc, int n) {
    int i = blockIdx.x * blockDim.x + threadIdx.x;
    if (i >= n * DIM) return;
    int nid = i / DIM, d = i % DIM;
    float v;
    if (x[(size_t)nid * FDIM] == 2.0f) {
        v = bup[d];
        for (int t = 0; t < BTY; ++t) v += oe[nid * BTY + t] * wup[d * BTY + t];
    } else v = state[i];
    hc[i] = v;
}

// one wave per edge: me = hc[src] @ Mt[argmax(edge_attr)]; scatter to agg[dst]
__global__ __launch_bounds__(256) void fg_edge(
    const int* __restrict__ src, const int* __restrict__ dst,
    const float* __restrict__ ea, const float* __restrict__ hc,
    const float* __restrict__ Mt, float* __restrict__ agg, int e)
{
    int w = (blockIdx.x * blockDim.x + threadIdx.x) >> 5;
    int lane = threadIdx.x & 31;
    if (w >= e) return;
    int s = src[w], d = dst[w];
    const float* a = ea + (size_t)w * BTY;
    int bond = 0; float best = a[0];
    for (int t = 1; t < BTY; ++t) if (a[t] > best) { best = a[t]; bond = t; }
    const float* __restrict__ Mm = Mt + (size_t)bond * DIM * DIM + 2 * lane;
    const float* __restrict__ h  = hc + (size_t)s * DIM;
    float a0 = 0.0f, a1 = 0.0f;
#pragma unroll 8
    for (int i = 0; i < DIM; ++i) {
        float hv = h[i];
        f32x2 mv = *(const f32x2*)(Mm + i * DIM);
        a0 += hv * mv[0];
        a1 += hv * mv[1];
    }
    atomicAdd(&agg[(size_t)d * DIM + 2 * lane],     a0);
    atomicAdd(&agg[(size_t)d * DIM + 2 * lane + 1], a1);
}

// msg_f -> WDown -> weight_e -> linWe -> add to out_edges -> log_softmax (4 classes)
__global__ void final_kernel(const float* __restrict__ aggF, const float* __restrict__ invdeg,
                             const float* __restrict__ wdown, const float* __restrict__ bdown,
                             const float* __restrict__ we, const float* __restrict__ linWe,
                             const float* __restrict__ linbe, const float* __restrict__ oe,
                             float* __restrict__ out, int n)
{
    int nid = blockIdx.x * blockDim.x + threadIdx.x;
    if (nid >= n) return;
    float inv = invdeg[nid];
    float mte[BTY] = {0, 0, 0, 0};
    for (int d = 0; d < DIM; ++d) {
        float mf = fmaxf(aggF[(size_t)nid * DIM + d] * inv, 0.0f);
        for (int t = 0; t < BTY; ++t) mte[t] += mf * wdown[t * DIM + d];
    }
    float q[BTY], o[BTY];
    for (int t = 0; t < BTY; ++t) q[t] = we[t] * (mte[t] + bdown[t]);
    for (int t = 0; t < BTY; ++t) {
        float s = linbe[t];
        for (int u = 0; u < BTY; ++u) s += q[u] * linWe[t * BTY + u];
        o[t] = oe[nid * BTY + t] + fmaxf(s, 0.0f);
    }
    float mx = fmaxf(fmaxf(o[0], o[1]), fmaxf(o[2], o[3]));
    float lse = 0.0f;
    for (int t = 0; t < BTY; ++t) lse += expf(o[t] - mx);
    lse = logf(lse);
    for (int t = 0; t < BTY; ++t) out[nid * BTY + t] = o[t] - mx - lse;
}

// ---------------------------------------------------------------------------
static inline dim3 gemm_grid(int M, int N, int NT) {
    int tilesM = (M + 15) / 16;
    int grpN   = (N + NT * 16 - 1) / (NT * 16);
    int waves  = tilesM * grpN;
    return dim3((waves + 7) / 8);   // 256 threads = 8 wave32
}
#define ELW(n) dim3(((n) + 255) / 256), dim3(256)

extern "C" void kernel_launch(void* const* d_in, const int* in_sizes, int n_in,
                              void* d_out, int out_size, void* d_ws, size_t ws_size,
                              hipStream_t stream) {
    const float* x     = (const float*)d_in[0];
    const int*   ei    = (const int*)  d_in[1];
    const float* ea    = (const float*)d_in[2];
    const float* W0    = (const float*)d_in[3];
    const float* b0    = (const float*)d_in[4];
    const float* nnW1  = (const float*)d_in[5];
    const float* nnb1  = (const float*)d_in[6];
    const float* nnW2  = (const float*)d_in[7];
    const float* nnb2  = (const float*)d_in[8];
    const float* root  = (const float*)d_in[9];
    const float* convb = (const float*)d_in[10];
    const float* Wih   = (const float*)d_in[11];
    const float* Whh   = (const float*)d_in[12];
    const float* bih   = (const float*)d_in[13];
    const float* bhh   = (const float*)d_in[14];
    const float* W1l   = (const float*)d_in[15];
    const float* b1l   = (const float*)d_in[16];
    const float* WUp   = (const float*)d_in[17];
    const float* bUp   = (const float*)d_in[18];
    const float* WDown = (const float*)d_in[19];
    const float* bDown = (const float*)d_in[20];
    const float* U     = (const float*)d_in[21];
    const float* V     = (const float*)d_in[22];
    /* weight(23), linW(24), linb(25): dead code w.r.t. reference output */
    const float* we    = (const float*)d_in[26];
    const float* linWe = (const float*)d_in[27];
    const float* linbe = (const float*)d_in[28];

    const int* src = ei;
    const int* dst = ei + NE;

    // workspace carve (256B aligned)
    char* w = (char*)d_ws;
    auto take = [&](size_t bytes) { void* p = w; w += (bytes + 255) & ~(size_t)255; return p; };
    float*  state  = (float*)take((size_t)NN * DIM * 4);
    float*  gi     = (float*)take((size_t)NN * 3 * DIM * 4);
    float*  gh     = (float*)take((size_t)NN * 3 * DIM * 4);
    float*  agg    = (float*)take((size_t)NN * DIM * 4);
    float*  deg    = (float*)take((size_t)NN * 4);
    float*  invdeg = (float*)take((size_t)NN * 4);
    float*  hid    = (float*)take((size_t)NE * 32 * 4);
    float*  Mt     = (float*)take((size_t)BTY * DIM * DIM * 4);
    float*  oed    = (float*)take((size_t)NN * BTY * 4);
    float*  hc     = (float*)take((size_t)NN * DIM * 4);
    __bf16* P      = (__bf16*)take((size_t)NN * PCOLS * 2);   // 33 MB, L2-resident
    __bf16* x_bf   = (__bf16*)take((size_t)NN * KPAD * 2);
    __bf16* st_bf  = (__bf16*)take((size_t)NN * DIM * 2);
    __bf16* m_bf   = (__bf16*)take((size_t)NN * DIM * 2);
    __bf16* W0b    = (__bf16*)take((size_t)DIM * KPAD * 2);
    __bf16* rootTb = (__bf16*)take((size_t)DIM * DIM * 2);
    __bf16* tmatTb = (__bf16*)take((size_t)PCOLS * DIM * 2);
    __bf16* Wihb   = (__bf16*)take((size_t)3 * DIM * DIM * 2);
    __bf16* Whhb   = (__bf16*)take((size_t)3 * DIM * DIM * 2);
    __bf16* W1lb   = (__bf16*)take((size_t)BTY * DIM * 2);
    __bf16* Vb     = (__bf16*)take((size_t)BTY * DIM * RNK * 2);
    __bf16* Utb    = (__bf16*)take((size_t)BTY * DIM * RNK * 2);

    // --- weight preparation (bf16, B^T layouts, K padded to 32 where needed)
    pad32_bf16<<<ELW(NN * KPAD), 0, stream>>>(x, x_bf, NN);
    pad32_bf16<<<ELW(DIM * KPAD), 0, stream>>>(W0, W0b, DIM);
    rootT_bf16<<<ELW(DIM * DIM), 0, stream>>>(root, rootTb);
    tmatT_bf16<<<ELW(PCOLS * DIM), 0, stream>>>(nnW2, nnb2, tmatTb);
    cvt_bf16<<<ELW(3 * DIM * DIM), 0, stream>>>(Wih, Wihb, 3 * DIM * DIM);
    cvt_bf16<<<ELW(3 * DIM * DIM), 0, stream>>>(Whh, Whhb, 3 * DIM * DIM);
    cvt_bf16<<<ELW(BTY * DIM), 0, stream>>>(W1l, W1lb, BTY * DIM);
    cvt_bf16<<<ELW(BTY * DIM * RNK), 0, stream>>>(V, Vb, BTY * DIM * RNK);
    uT_bf16<<<ELW(BTY * DIM * RNK), 0, stream>>>(U, Utb);

    // degree / inverse degree
    zero_f32<<<ELW(NN), 0, stream>>>(deg, NN);
    count_deg<<<ELW(NE), 0, stream>>>(dst, deg, NE);
    finalize_invdeg<<<ELW(NN), 0, stream>>>(invdeg, deg, NN);

    // state0 = relu(x @ W0^T + b0)   (K padded to 32)
    wmma_gemm<1><<<gemm_grid(NN, DIM, 1), 256, 0, stream>>>(
        x_bf, W0b, state, st_bf, b0, nullptr, nullptr,
        NN, DIM, KPAD, KPAD, KPAD, DIM, 1);

    hidden_kernel<<<ELW(NE * 32), 0, stream>>>(ea, nnW1, nnb1, hid, NE);

    for (int it = 0; it < 3; ++it) {
        // P = state @ Tmat^T   [8192,64]x[64,2112] -> bf16 (stays in L2)
        wmma_gemm<4><<<gemm_grid(NN, PCOLS, 4), 256, 0, stream>>>(
            st_bf, tmatTb, nullptr, P, nullptr, nullptr, nullptr,
            NN, PCOLS, DIM, DIM, DIM, PCOLS, 0);
        zero_f32<<<ELW(NN * DIM), 0, stream>>>(agg, NN * DIM);
        edge_msg<<<ELW(NE * 32), 0, stream>>>(src, dst, hid, P, agg, NE);
        // m = relu(state @ root + agg*invdeg + conv_b)  (bf16 shadow only)
        wmma_gemm<1><<<gemm_grid(NN, DIM, 1), 256, 0, stream>>>(
            st_bf, rootTb, nullptr, m_bf, convb, agg, invdeg,
            NN, DIM, DIM, DIM, DIM, DIM, 1);
        // gi = m @ Wih^T + bih ; gh = state @ Whh^T + bhh
        wmma_gemm<4><<<gemm_grid(NN, 3 * DIM, 4), 256, 0, stream>>>(
            m_bf, Wihb, gi, nullptr, bih, nullptr, nullptr,
            NN, 3 * DIM, DIM, DIM, DIM, 3 * DIM, 0);
        wmma_gemm<4><<<gemm_grid(NN, 3 * DIM, 4), 256, 0, stream>>>(
            st_bf, Whhb, gh, nullptr, bhh, nullptr, nullptr,
            NN, 3 * DIM, DIM, DIM, DIM, 3 * DIM, 0);
        gru_kernel<<<ELW(NN * DIM), 0, stream>>>(gi, gh, state, st_bf, NN);
    }

    // out_edges = relu(state @ W1l^T + b1l)   [8192,4]
    wmma_gemm<1><<<gemm_grid(NN, BTY, 1), 256, 0, stream>>>(
        st_bf, W1lb, oed, nullptr, b1l, nullptr, nullptr,
        NN, BTY, DIM, DIM, DIM, BTY, 1);
    combine_kernel<<<ELW(NN * DIM), 0, stream>>>(x, oed, WUp, bUp, state, hc, NN);

    // Mt[t] = V[t] @ U[t]   ([64,512]x[512,64]) collapses rank-512 factorization
    for (int t = 0; t < BTY; ++t)
        wmma_gemm<1><<<gemm_grid(DIM, DIM, 1), 256, 0, stream>>>(
            Vb + (size_t)t * DIM * RNK, Utb + (size_t)t * DIM * RNK,
            Mt + (size_t)t * DIM * DIM, nullptr, nullptr, nullptr, nullptr,
            DIM, DIM, RNK, RNK, RNK, DIM, 0);

    zero_f32<<<ELW(NN * DIM), 0, stream>>>(agg, NN * DIM);
    fg_edge<<<ELW(NE * 32), 0, stream>>>(src, dst, ea, hc, Mt, agg, NE);

    final_kernel<<<ELW(NN), 0, stream>>>(
        agg, invdeg, WDown, bDown, we, linWe, linbe, oed, (float*)d_out, NN);
}
